// Decode_Layer_69801808495348
// MI455X (gfx1250) — compile-verified, matrix operand order checked by reading, not run on hardware
//
#include <hip/hip_runtime.h>

// ---------------------------------------------------------------------------
// Types
// ---------------------------------------------------------------------------
typedef _Float16 half8  __attribute__((ext_vector_type(8)));
typedef _Float16 half16 __attribute__((ext_vector_type(16)));
typedef float    float8 __attribute__((ext_vector_type(8)));
typedef int      int4v  __attribute__((ext_vector_type(4)));

union FragU { half16 v; half8 h[2]; };

__device__ __forceinline__ float8 fzero8() {
    float8 z = {0.f,0.f,0.f,0.f,0.f,0.f,0.f,0.f};
    return z;
}

// WMMA wrapper: D = A(16x32 f16) x B(32x16 f16) + C(16x16 f32)
__device__ __forceinline__ float8 wmma_f16(half16 a, half16 b, float8 c) {
    return __builtin_amdgcn_wmma_f32_16x16x32_f16(
        /*neg_a=*/false, a, /*neg_b=*/false, b,
        /*c_mod=*/(short)0, c, /*reuse_a=*/false, /*reuse_b=*/false);
}

// ---------------------------------------------------------------------------
// CDNA5 async global->LDS copies (ASYNCcnt path), guarded so the file
// compiles on toolchains lacking the builtins (falls back to load+ds_store).
// Builtin signature (from compiler diagnostic): arg0 = int4 in AS(1)
// ("__device__"), arg1 = LDS dest (AS(3), "__shared__"), then imm offset, cpol.
// ---------------------------------------------------------------------------
#if defined(__gfx1250__) && \
    __has_builtin(__builtin_amdgcn_global_load_async_to_lds_b128) && \
    __has_builtin(__builtin_amdgcn_s_wait_asynccnt)
#define USE_ASYNC_LDS 1
#else
#define USE_ASYNC_LDS 0
#endif

#if USE_ASYNC_LDS
typedef __attribute__((address_space(1))) int4v as1_int4;
typedef __attribute__((address_space(3))) int4v as3_int4;

__device__ __forceinline__ void async_cp16(const _Float16* gsrc, _Float16* ldst) {
    // generic->AS casts via uintptr_t: AS(3) ptr = low 32 bits (LDS offset),
    // AS(1) ptr = flat VA.
    __builtin_amdgcn_global_load_async_to_lds_b128(
        (as1_int4*)(uintptr_t)gsrc,
        (as3_int4*)(uintptr_t)ldst,
        0, 0);
}
__device__ __forceinline__ void wait_async() {
    __builtin_amdgcn_s_wait_asynccnt(0);
}
#else
__device__ __forceinline__ void wait_async() {}
#endif

// Copy 8 halves global -> LDS (async when available).
__device__ __forceinline__ void cp16(const _Float16* gsrc, _Float16* ldst) {
#if USE_ASYNC_LDS
    async_cp16(gsrc, ldst);
#else
    *(half8*)ldst = *(const half8*)gsrc;
#endif
}

// Load a 16x32 f16 fragment from an LDS tile.
// A-fragment: lane&15 = row (M). B-fragment (tile stored [N][K]): lane&15 = col (N).
// 16-bit 16x32 layout: lanes 0-15 hold K halves {0..7,16..23},
// lanes 16-31 hold {8..15,24..31} -> two contiguous 16B LDS loads.
__device__ __forceinline__ half16 ldsFrag(const _Float16* tile, int stride,
                                          int lane, int kOff) {
    const _Float16* p = tile + (lane & 15) * stride + kOff + ((lane & 16) ? 8 : 0);
    FragU f;
    f.h[0] = *(const half8*)(p);
    f.h[1] = *(const half8*)(p + 16);
    return f.v;
}

// ---------------------------------------------------------------------------
// Weight convert + transpose:  in f32 [K][N] -> out f16 [N][K]
// ---------------------------------------------------------------------------
__global__ __launch_bounds__(256)
void transpose_cvt_kernel(const float* __restrict__ in, _Float16* __restrict__ out,
                          int K, int N) {
    __shared__ float t[32][33];
    const int n0 = blockIdx.x * 32, k0 = blockIdx.y * 32;
    for (int i = threadIdx.y; i < 32; i += 8)
        t[i][threadIdx.x] = in[(size_t)(k0 + i) * N + n0 + threadIdx.x];
    __syncthreads();
    for (int i = threadIdx.y; i < 32; i += 8)
        out[(size_t)(n0 + i) * K + k0 + threadIdx.x] = (_Float16)t[threadIdx.x][i];
}

// ---------------------------------------------------------------------------
// Elementwise f32 -> f16
// ---------------------------------------------------------------------------
__global__ __launch_bounds__(256)
void cvt_f16_kernel(const float* __restrict__ in, _Float16* __restrict__ out, int n) {
    int i = blockIdx.x * 256 + threadIdx.x;
    if (i < n) out[i] = (_Float16)in[i];
}

// ---------------------------------------------------------------------------
// RMSNorm: per row of [M, D] f32, write f16 (GEMM input)
// ---------------------------------------------------------------------------
__global__ __launch_bounds__(256)
void rmsnorm_f16_kernel(const float* __restrict__ x, const float* __restrict__ g,
                        _Float16* __restrict__ out, int D) {
    const int row = blockIdx.x;
    const float* xr = x + (size_t)row * D;
    float ss = 0.f;
    for (int c = threadIdx.x; c < D; c += 256) { float v = xr[c]; ss += v * v; }
    #pragma unroll
    for (int off = 16; off > 0; off >>= 1) ss += __shfl_xor(ss, off, 32);
    __shared__ float red[8];
    __shared__ float sinv;
    if ((threadIdx.x & 31) == 0) red[threadIdx.x >> 5] = ss;
    __syncthreads();
    if (threadIdx.x == 0) {
        float t = 0.f;
        #pragma unroll
        for (int i = 0; i < 8; ++i) t += red[i];
        sinv = rsqrtf(t / (float)D + 1e-6f);
    }
    __syncthreads();
    const float inv = sinv;
    _Float16* orow = out + (size_t)row * D;
    for (int c = threadIdx.x; c < D; c += 256)
        orow[c] = (_Float16)(xr[c] * inv * g[c]);
}

// ---------------------------------------------------------------------------
// WMMA GEMM:  C[M,N] = A[M,K](f16) * Bt[N,K](f16)^T, epilogue variants.
// Block tile 128x128, BK=32, 256 threads (8 waves), wave tile 32x64.
// Double-buffered LDS, async global->LDS staging, one barrier per K-step.
// ---------------------------------------------------------------------------
__global__ __launch_bounds__(256)
void wmma_gemm_kernel(const _Float16* __restrict__ A, const _Float16* __restrict__ Bt,
                      float* __restrict__ Cf32, _Float16* __restrict__ Cf16,
                      const float* __restrict__ residual,
                      int M, int N, int K, float scale, int relu) {
    __shared__ __align__(16) _Float16 As[2][128][40];   // +8 halves pad
    __shared__ __align__(16) _Float16 Bs[2][128][40];

    const int tid  = threadIdx.x;
    const int lane = tid & 31;
    const int wave = tid >> 5;
    const int wm   = wave >> 1;      // 0..3  (M direction)
    const int wn   = wave & 1;       // 0..1  (N direction)
    const int m0   = blockIdx.y * 128;
    const int n0   = blockIdx.x * 128;

    float8 acc[2][4];
    #pragma unroll
    for (int i = 0; i < 2; ++i)
        #pragma unroll
        for (int j = 0; j < 4; ++j) acc[i][j] = fzero8();

    // Stage one 128x32 A/B tile pair into buffer `buf` (512 16B chunks).
    auto stage = [&](int buf, int k0) {
        #pragma unroll
        for (int it = 0; it < 2; ++it) {
            int c = tid + it * 256;
            int r = c >> 2, ch = (c & 3) * 8;
            cp16(&A [(size_t)(m0 + r) * K + k0 + ch], &As[buf][r][ch]);
            cp16(&Bt[(size_t)(n0 + r) * K + k0 + ch], &Bs[buf][r][ch]);
        }
    };

    stage(0, 0);
    for (int k0 = 0; k0 < K; k0 += 32) {
        const int cur = (k0 >> 5) & 1;
        wait_async();           // own async copies done -> barrier makes all visible
        __syncthreads();
        if (k0 + 32 < K) {
            stage(cur ^ 1, k0 + 32);          // overlap next tile with compute
            __builtin_prefetch(&A [(size_t)(m0 + (tid >> 1)) * K + k0 + 64], 0, 1);
            __builtin_prefetch(&Bt[(size_t)(n0 + (tid >> 1)) * K + k0 + 64], 0, 1);
        }

        half16 af[2], bf[4];
        #pragma unroll
        for (int i = 0; i < 2; ++i)
            af[i] = ldsFrag(&As[cur][wm * 32 + i * 16][0], 40, lane, 0);
        #pragma unroll
        for (int j = 0; j < 4; ++j)
            bf[j] = ldsFrag(&Bs[cur][wn * 64 + j * 16][0], 40, lane, 0);
        #pragma unroll
        for (int i = 0; i < 2; ++i)
            #pragma unroll
            for (int j = 0; j < 4; ++j)
                acc[i][j] = wmma_f16(af[i], bf[j], acc[i][j]);
    }

    // Epilogue. C layout: VGPR r -> row (r + 8*(lane>=16)), col = lane&15.
    const int rowOff = ((lane >> 4) << 3);
    #pragma unroll
    for (int i = 0; i < 2; ++i) {
        #pragma unroll
        for (int j = 0; j < 4; ++j) {
            const int row = m0 + wm * 32 + i * 16 + rowOff;
            const int col = n0 + wn * 64 + j * 16 + (lane & 15);
            #pragma unroll
            for (int r = 0; r < 8; ++r) {
                float v = acc[i][j][r] * scale;
                if (relu) v = fmaxf(v, 0.f);
                const size_t idx = (size_t)(row + r) * N + col;
                if (residual) v += residual[idx];
                if (Cf16) Cf16[idx] = (_Float16)v;
                else      Cf32[idx] = v;
            }
        }
    }
}

// ---------------------------------------------------------------------------
// Fused flash attention (WMMA for QK^T and PV, online softmax).
// One block = (b, h, 64 q rows), 4 waves, each wave owns 16 q rows.
// Q is pre-scaled by 1/sqrt(hd). bias is [Lq, Lk] f32 (mask or pos-emb).
// Q/K/V/O layout: [B*L, D] f16, head h at columns h*64.
// Double-buffered K / V^T tiles, one barrier per kv iteration.
// ---------------------------------------------------------------------------
__global__ __launch_bounds__(128)
void flash_attn_kernel(const _Float16* __restrict__ Q, const _Float16* __restrict__ Kv,
                       const _Float16* __restrict__ Vv, const float* __restrict__ bias,
                       _Float16* __restrict__ O, int Lq, int Lk, int ld, int causal) {
    __shared__ __align__(16) _Float16 Qs [64][72];
    __shared__ __align__(16) _Float16 Ks [2][32][72];
    __shared__ __align__(16) _Float16 Vts[2][64][40];      // V transposed: [d][kv]
    __shared__ __align__(16) _Float16 Ps [4][16][40];      // per-wave P tile [q][kv]

    const int tid  = threadIdx.x;
    const int lane = tid & 31;
    const int w    = tid >> 5;
    const int qb   = blockIdx.x * 64;
    const int h    = blockIdx.y;
    const int b    = blockIdx.z;

    const _Float16* Qp = Q  + ((size_t)(b * Lq + qb)) * ld + h * 64;
    const _Float16* Kp = Kv + ((size_t)(b * Lk))      * ld + h * 64;
    const _Float16* Vp = Vv + ((size_t)(b * Lk))      * ld + h * 64;

    // Stage the 64x64 Q tile once (async when available).
    #pragma unroll
    for (int it = 0; it < 4; ++it) {
        int c = tid + it * 128;
        int r = c >> 3, ch = (c & 7) * 8;
        cp16(&Qp[(size_t)r * ld + ch], &Qs[r][ch]);
    }

    // Stage K tile [32][64] (async) and V^T tile [64][32] (register transpose).
    auto stageKV = [&](int buf, int kv0) {
        #pragma unroll
        for (int it = 0; it < 2; ++it) {
            int c = tid + it * 128;
            int r = c >> 3, ch = (c & 7) * 8;
            cp16(&Kp[(size_t)(kv0 + r) * ld + ch], &Ks[buf][r][ch]);
            half8 v = *(const half8*)&Vp[(size_t)(kv0 + r) * ld + ch];
            #pragma unroll
            for (int e = 0; e < 8; ++e) Vts[buf][ch + e][r] = v[e];
        }
    };

    float m[8], l[8];
    float8 o[4];
    #pragma unroll
    for (int r = 0; r < 8; ++r) { m[r] = -3.0e38f; l[r] = 0.f; }
    #pragma unroll
    for (int j = 0; j < 4; ++j) o[j] = fzero8();

    const int rowq = qb + w * 16 + ((lane >> 4) << 3);   // + r = q position
    const int kend = causal ? (qb + 64) : Lk;

    stageKV(0, 0);
    for (int kv0 = 0; kv0 < kend; kv0 += 32) {
        const int cur = (kv0 >> 5) & 1;
        wait_async();
        __syncthreads();
        if (kv0 + 32 < kend) stageKV(cur ^ 1, kv0 + 32);

        // S (16q x 32kv) = Q K^T : two k=32 WMMA steps over hd=64, two N tiles.
        float8 s0 = fzero8(), s1 = fzero8();
        #pragma unroll
        for (int d0 = 0; d0 < 64; d0 += 32) {
            half16 aq = ldsFrag(&Qs[w * 16][0],       72, lane, d0);
            half16 b0 = ldsFrag(&Ks[cur][0][0],       72, lane, d0);
            half16 b1 = ldsFrag(&Ks[cur][16][0],      72, lane, d0);
            s0 = wmma_f16(aq, b0, s0);
            s1 = wmma_f16(aq, b1, s1);
        }

        // Bias + online softmax (rows live across 16-lane groups).
        #pragma unroll
        for (int r = 0; r < 8; ++r) {
            const size_t brow = (size_t)(rowq + r) * Lk + kv0;
            float v0 = s0[r] + bias[brow + (lane & 15)];
            float v1 = s1[r] + bias[brow + 16 + (lane & 15)];
            float mx = fmaxf(v0, v1);
            #pragma unroll
            for (int off = 1; off < 16; off <<= 1) mx = fmaxf(mx, __shfl_xor(mx, off, 32));
            const float mn    = fmaxf(m[r], mx);
            const float alpha = __expf(m[r] - mn);
            const float p0    = __expf(v0 - mn);
            const float p1    = __expf(v1 - mn);
            float ps = p0 + p1;
            #pragma unroll
            for (int off = 1; off < 16; off <<= 1) ps += __shfl_xor(ps, off, 32);
            m[r] = mn;
            l[r] = l[r] * alpha + ps;
            #pragma unroll
            for (int j = 0; j < 4; ++j) o[j][r] *= alpha;
            const int pr = r + ((lane >> 4) << 3);
            Ps[w][pr][(lane & 15)]      = (_Float16)p0;
            Ps[w][pr][(lane & 15) + 16] = (_Float16)p1;
        }

        // O(16q x 64d) += P(16x32) @ V(32x64). Ps is wave-private: per-wave
        // LDS ordering + compiler s_wait_dscnt suffice, no barrier needed.
        half16 ap = ldsFrag(&Ps[w][0][0], 40, lane, 0);
        #pragma unroll
        for (int j = 0; j < 4; ++j) {
            half16 bv = ldsFrag(&Vts[cur][j * 16][0], 40, lane, 0);
            o[j] = wmma_f16(ap, bv, o[j]);
        }
    }

    // Normalize and store O as f16.
    #pragma unroll
    for (int j = 0; j < 4; ++j) {
        #pragma unroll
        for (int r = 0; r < 8; ++r) {
            const float val = o[j][r] / l[r];
            const size_t idx = ((size_t)(b * Lq + rowq + r)) * ld + h * 64 + j * 16 + (lane & 15);
            O[idx] = (_Float16)val;
        }
    }
}

// ---------------------------------------------------------------------------
// Host orchestration
// ---------------------------------------------------------------------------
static inline void gemm(hipStream_t s, const _Float16* A, const _Float16* Bt,
                        float* cf32, _Float16* cf16, const float* res,
                        int M, int N, int K, float scale, int relu) {
    dim3 g(N / 128, M / 128), b(256);
    wmma_gemm_kernel<<<g, b, 0, s>>>(A, Bt, cf32, cf16, res, M, N, K, scale, relu);
}

extern "C" void kernel_launch(void* const* d_in, const int* in_sizes, int n_in,
                              void* d_out, int out_size, void* d_ws, size_t ws_size,
                              hipStream_t stream) {
    (void)in_sizes; (void)n_in; (void)out_size; (void)ws_size;
    const int D = 1024, H = 16, FFN = 4096, B = 2, LT = 2048, LS = 2048;
    const int M = B * LT;                 // 4096 rows
    const float qscale = 0.125f;          // 1/sqrt(64)

    const float* x    = (const float*)d_in[0];
    const float* mem  = (const float*)d_in[1];
    const float* pemb = (const float*)d_in[2];
    const float* cmsk = (const float*)d_in[3];
    const float* g1   = (const float*)d_in[4];
    const float* Wq1  = (const float*)d_in[5];
    const float* Wk1  = (const float*)d_in[6];
    const float* Wv1  = (const float*)d_in[7];
    const float* Wo1  = (const float*)d_in[8];
    const float* g2   = (const float*)d_in[9];
    const float* Wq2  = (const float*)d_in[10];
    const float* Wk2  = (const float*)d_in[11];
    const float* Wv2  = (const float*)d_in[12];
    const float* Wo2  = (const float*)d_in[13];
    const float* g3   = (const float*)d_in[14];
    const float* W1   = (const float*)d_in[15];
    const float* W2   = (const float*)d_in[16];

    char* ws = (char*)d_ws;
    size_t off = 0;
    auto alloc = [&](size_t bytes) -> char* {
        char* p = ws + off;
        off = (off + bytes + 255) & ~(size_t)255;
        return p;
    };
    const size_t szDD  = (size_t)D * D * sizeof(_Float16);
    const size_t szAct = (size_t)M * D * sizeof(_Float16);

    _Float16* Wq1t = (_Float16*)alloc(szDD);
    _Float16* Wk1t = (_Float16*)alloc(szDD);
    _Float16* Wv1t = (_Float16*)alloc(szDD);
    _Float16* Wo1t = (_Float16*)alloc(szDD);
    _Float16* Wq2t = (_Float16*)alloc(szDD);
    _Float16* Wk2t = (_Float16*)alloc(szDD);
    _Float16* Wv2t = (_Float16*)alloc(szDD);
    _Float16* Wo2t = (_Float16*)alloc(szDD);
    _Float16* W1t  = (_Float16*)alloc((size_t)D * FFN * sizeof(_Float16));
    _Float16* W2t  = (_Float16*)alloc((size_t)D * FFN * sizeof(_Float16));
    _Float16* h16  = (_Float16*)alloc(szAct);
    _Float16* m16  = (_Float16*)alloc(szAct);
    _Float16* Qb   = (_Float16*)alloc(szAct);
    _Float16* Kb   = (_Float16*)alloc(szAct);
    _Float16* Vb   = (_Float16*)alloc(szAct);
    _Float16* Ob   = (_Float16*)alloc(szAct);
    _Float16* t16  = (_Float16*)alloc((size_t)M * FFN * sizeof(_Float16));
    float*    x1   = (float*)alloc((size_t)M * D * sizeof(float));
    float*    x2   = (float*)alloc((size_t)M * D * sizeof(float));

    // --- Weight conversion (f32 [K][N] -> f16 [N][K]) ---
    {
        dim3 b(32, 8);
        dim3 gDD(D / 32, D / 32);
        transpose_cvt_kernel<<<gDD, b, 0, stream>>>(Wq1, Wq1t, D, D);
        transpose_cvt_kernel<<<gDD, b, 0, stream>>>(Wk1, Wk1t, D, D);
        transpose_cvt_kernel<<<gDD, b, 0, stream>>>(Wv1, Wv1t, D, D);
        transpose_cvt_kernel<<<gDD, b, 0, stream>>>(Wo1, Wo1t, D, D);
        transpose_cvt_kernel<<<gDD, b, 0, stream>>>(Wq2, Wq2t, D, D);
        transpose_cvt_kernel<<<gDD, b, 0, stream>>>(Wk2, Wk2t, D, D);
        transpose_cvt_kernel<<<gDD, b, 0, stream>>>(Wv2, Wv2t, D, D);
        transpose_cvt_kernel<<<gDD, b, 0, stream>>>(Wo2, Wo2t, D, D);
        transpose_cvt_kernel<<<dim3(FFN / 32, D / 32), b, 0, stream>>>(W1, W1t, D, FFN);
        transpose_cvt_kernel<<<dim3(D / 32, FFN / 32), b, 0, stream>>>(W2, W2t, FFN, D);
    }
    // memory -> f16 (input to K2/V2 projections)
    {
        int n = M * D;
        cvt_f16_kernel<<<(n + 255) / 256, 256, 0, stream>>>(mem, m16, n);
    }

    dim3 fgrid(LT / 64, H, B), fblk(128);

    // --- Self-attention block ---
    rmsnorm_f16_kernel<<<M, 256, 0, stream>>>(x, g1, h16, D);
    gemm(stream, h16, Wq1t, nullptr, Qb, nullptr, M, D, D, qscale, 0);
    gemm(stream, h16, Wk1t, nullptr, Kb, nullptr, M, D, D, 1.0f, 0);
    gemm(stream, h16, Wv1t, nullptr, Vb, nullptr, M, D, D, 1.0f, 0);
    flash_attn_kernel<<<fgrid, fblk, 0, stream>>>(Qb, Kb, Vb, cmsk, Ob, LT, LT, D, 1);
    gemm(stream, Ob, Wo1t, x1, nullptr, x, M, D, D, 1.0f, 0);   // x1 = O@Wo1 + x

    // --- Cross-attention block ---
    rmsnorm_f16_kernel<<<M, 256, 0, stream>>>(x1, g2, h16, D);
    gemm(stream, h16, Wq2t, nullptr, Qb, nullptr, M, D, D, qscale, 0);
    gemm(stream, m16, Wk2t, nullptr, Kb, nullptr, M, D, D, 1.0f, 0);
    gemm(stream, m16, Wv2t, nullptr, Vb, nullptr, M, D, D, 1.0f, 0);
    flash_attn_kernel<<<fgrid, fblk, 0, stream>>>(Qb, Kb, Vb, pemb, Ob, LT, LS, D, 0);
    gemm(stream, Ob, Wo2t, x2, nullptr, x1, M, D, D, 1.0f, 0);  // x2 = O@Wo2 + x1

    // --- FFN block ---
    rmsnorm_f16_kernel<<<M, 256, 0, stream>>>(x2, g3, h16, D);
    gemm(stream, h16, W1t, nullptr, t16, nullptr, M, FFN, D, 1.0f, 1);      // relu
    gemm(stream, t16, W2t, (float*)d_out, nullptr, x2, M, D, FFN, 1.0f, 0); // + x2
}
